// HybridAttention_42932493091272
// MI455X (gfx1250) — matrix operand order repaired and checked by reading, not compiled, not run on hardware
//
#include <hip/hip_runtime.h>
#include <hip/hip_bf16.h>

// ---------------- problem constants ----------------
#define Bb   4
#define Tt   2048
#define Ff   256
#define Hh   4
#define Cc   78
#define DKk  64
#define HALFc 38
#define NEGF (-3.0e38f)
#define NWREAL (Hh * Cc)     // 312
#define NWPAD  384           // padded to 3 x 128 tiles

typedef __attribute__((ext_vector_type(16))) __bf16 v16bf;
typedef __attribute__((ext_vector_type(8)))  __bf16 v8bf;
typedef __attribute__((ext_vector_type(4)))  __bf16 v4bf;
typedef __attribute__((ext_vector_type(8)))  float  v8f;
typedef __attribute__((ext_vector_type(4)))  float  v4f;

#define DEV __device__ __forceinline__

DEV v16bf cat8(v8bf lo, v8bf hi) {
  return __builtin_shufflevector(lo, hi, 0,1,2,3,4,5,6,7,8,9,10,11,12,13,14,15);
}

// load 16 consecutive fp32, convert to bf16 (B-fragment helper)
DEV v16bf load16f_bf(const float* __restrict__ p) {
  v4f f0 = *(const v4f*)p;
  v4f f1 = *(const v4f*)(p + 4);
  v4f f2 = *(const v4f*)(p + 8);
  v4f f3 = *(const v4f*)(p + 12);
  v16bf r;
  r[0]=(__bf16)f0.x; r[1]=(__bf16)f0.y; r[2]=(__bf16)f0.z; r[3]=(__bf16)f0.w;
  r[4]=(__bf16)f1.x; r[5]=(__bf16)f1.y; r[6]=(__bf16)f1.z; r[7]=(__bf16)f1.w;
  r[8]=(__bf16)f2.x; r[9]=(__bf16)f2.y; r[10]=(__bf16)f2.z; r[11]=(__bf16)f2.w;
  r[12]=(__bf16)f3.x; r[13]=(__bf16)f3.y; r[14]=(__bf16)f3.z; r[15]=(__bf16)f3.w;
  return r;
}

// A-fragment from global: elems 0..7 = p[0..7], elems 8..15 = p[16..23]
DEV v16bf loadA16f_bf(const float* __restrict__ p) {
  v4f f0 = *(const v4f*)p;
  v4f f1 = *(const v4f*)(p + 4);
  v4f f2 = *(const v4f*)(p + 16);
  v4f f3 = *(const v4f*)(p + 20);
  v16bf r;
  r[0]=(__bf16)f0.x; r[1]=(__bf16)f0.y; r[2]=(__bf16)f0.z; r[3]=(__bf16)f0.w;
  r[4]=(__bf16)f1.x; r[5]=(__bf16)f1.y; r[6]=(__bf16)f1.z; r[7]=(__bf16)f1.w;
  r[8]=(__bf16)f2.x; r[9]=(__bf16)f2.y; r[10]=(__bf16)f2.z; r[11]=(__bf16)f2.w;
  r[12]=(__bf16)f3.x; r[13]=(__bf16)f3.y; r[14]=(__bf16)f3.z; r[15]=(__bf16)f3.w;
  return r;
}

// =====================================================================
// Zero-pad w2 (256 x 312) -> w2p (256 x 384) so the band-weights GEMM
// runs with exact 128-wide tiles and no bounds checks anywhere.
// =====================================================================
__global__ __launch_bounds__(256)
void pad_w2(const float* __restrict__ w2, float* __restrict__ w2p) {
  const int idx = blockIdx.x * 256 + threadIdx.x;   // 256*384 total
  const int k = idx / NWPAD, n = idx % NWPAD;
  const int nc = (n < NWREAL) ? n : (NWREAL - 1);
  float f = w2[(size_t)k * NWREAL + nc];
  w2p[idx] = (n < NWREAL) ? f : 0.0f;
}

// =====================================================================
// Generic tiled GEMM: C[M,N] = act(A[M,K] @ W[K,N] + bias)
// block = 256 threads (8 waves). Block tile 64x128, K step 64,
// double-buffered LDS (one barrier per stage). Wave tile 16x64
// -> 8 WMMAs per wave per stage.  Requires M%64==0, N%128==0, K%64==0.
// =====================================================================
__global__ __launch_bounds__(256)
void gemm_bf16_wmma(const float* __restrict__ A, const float* __restrict__ W,
                    const float* __restrict__ bias, float* __restrict__ Cmat,
                    int M, int N, int K, int relu_out)
{
  constexpr int ST = 72;   // LDS row stride (bf16 elems): 144B = multiple of 16B
  __shared__ __align__(16) __bf16 la [2][64  * ST];   // A tiles  (64 rows x 64 k)
  __shared__ __align__(16) __bf16 lwt[2][128 * ST];   // W tiles TRANSPOSED (128 n x 64 k)

  const int tid   = threadIdx.x;
  const int wv    = tid >> 5;
  const int ln    = tid & 31;
  const int lmod  = ln & 15;
  const int lhalf = ln >> 4;
  const int m0 = blockIdx.x * 64;
  const int n0 = blockIdx.y * 128;
  const int wm = wv & 3;        // 4 wave-rows of 16
  const int wn = wv >> 2;       // 2 wave-cols of 64

  v8f acc[4] = {};

  auto load_stage = [&](int kt, int buf) {
    { // A: 64 rows x 64 k -> row-major bf16, vector LDS stores
      const int r  = tid >> 2;
      const int c0 = (tid & 3) * 16;
      const float* src = A + (size_t)(m0 + r) * K + kt + c0;
      if (kt + 64 < K) __builtin_prefetch(src + 64, 0, 1);
      __align__(16) __bf16 tmp[16];
      #pragma unroll
      for (int q = 0; q < 4; ++q) {
        v4f f = *(const v4f*)(src + q * 4);
        tmp[q*4+0]=(__bf16)f.x; tmp[q*4+1]=(__bf16)f.y;
        tmp[q*4+2]=(__bf16)f.z; tmp[q*4+3]=(__bf16)f.w;
      }
      *(v8bf*)&la[buf][r * ST + c0]     = *(const v8bf*)&tmp[0];
      *(v8bf*)&la[buf][r * ST + c0 + 8] = *(const v8bf*)&tmp[8];
    }
    { // W: 4 k-rows x 8 n-cols per thread, transposed store as ds_store_b64
      const int kp = (tid >> 4) * 4;      // 0..60
      const int c0 = (tid & 15) * 8;      // 0..120
      float rbuf[4][8];
      #pragma unroll
      for (int rr = 0; rr < 4; ++rr) {
        const float* src = W + (size_t)(kt + kp + rr) * N + n0 + c0;
        v4f f0 = *(const v4f*)src;
        v4f f1 = *(const v4f*)(src + 4);
        rbuf[rr][0]=f0.x; rbuf[rr][1]=f0.y; rbuf[rr][2]=f0.z; rbuf[rr][3]=f0.w;
        rbuf[rr][4]=f1.x; rbuf[rr][5]=f1.y; rbuf[rr][6]=f1.z; rbuf[rr][7]=f1.w;
      }
      #pragma unroll
      for (int j = 0; j < 8; ++j) {
        __align__(8) __bf16 q4[4];
        q4[0]=(__bf16)rbuf[0][j]; q4[1]=(__bf16)rbuf[1][j];
        q4[2]=(__bf16)rbuf[2][j]; q4[3]=(__bf16)rbuf[3][j];
        *(v4bf*)&lwt[buf][(c0 + j) * ST + kp] = *(const v4bf*)q4;
      }
    }
  };

  load_stage(0, 0);
  const int nst = K >> 6;
  for (int s = 0; s < nst; ++s) {
    __syncthreads();                        // stage-s LDS stores visible
    if (s + 1 < nst) load_stage((s + 1) * 64, (s + 1) & 1);
    const int buf = s & 1;
    const int arow = wm * 16 + lmod;
    #pragma unroll
    for (int kc = 0; kc < 2; ++kc) {
      // A fragment (16x32): lane half selects K 0-7/16-23 vs 8-15/24-31
      const int kb = kc * 32 + lhalf * 8;
      v8bf a_lo = *(const v8bf*)&la[buf][arow * ST + kb];
      v8bf a_hi = *(const v8bf*)&la[buf][arow * ST + kb + 16];
      v16bf afrag = cat8(a_lo, a_hi);
      // B fragments (32x16): lane half selects K 0-7/16-23 vs 8-15/24-31, col = lane%16
      const int bk = kc * 32 + lhalf * 8;
      #pragma unroll
      for (int t = 0; t < 4; ++t) {
        const int bcol = wn * 64 + t * 16 + lmod;
        v8bf blo = *(const v8bf*)&lwt[buf][bcol * ST + bk];
        v8bf bhi = *(const v8bf*)&lwt[buf][bcol * ST + bk + 16];
        v16bf bfrag = cat8(blo, bhi);
        acc[t] = __builtin_amdgcn_wmma_f32_16x16x32_bf16(false, afrag, false, bfrag,
                                                         (short)0, acc[t], false, false);
      }
    }
  }

  // epilogue (C layout: vgpr r -> row r + 8*half, col = lane%16); tiles exact
  #pragma unroll
  for (int t = 0; t < 4; ++t) {
    const int col = n0 + wn * 64 + t * 16 + lmod;
    const float bv = bias ? bias[col] : 0.0f;
    #pragma unroll
    for (int r = 0; r < 8; ++r) {
      const int row = m0 + wm * 16 + r + 8 * lhalf;
      float v = acc[t][r] + bv;
      if (relu_out) v = fmaxf(v, 0.0f);
      Cmat[(size_t)row * N + col] = v;
    }
  }
}

// =====================================================================
// Banded local-softmax attention (LDSA). One wave per (b,h,t).
// window of Cc=78 keys; 64-dim output per head.  Pure VALU (0.3 GF).
// =====================================================================
__global__ __launch_bounds__(256)
void ldsa_band(const float* __restrict__ Wgt,   // (B,T,NWPAD) padded
               const float* __restrict__ V3,    // (B,T,F)
               float* __restrict__ X)           // (B,T,F)
{
  __shared__ float lp[8][80];
  const int tid = threadIdx.x, wv = tid >> 5, ln = tid & 31;
  const int wid = blockIdx.x * 8 + wv;
  const int t = wid % Tt;
  const int h = (wid / Tt) % Hh;
  const int b = wid / (Tt * Hh);

  const float* wp = Wgt + ((size_t)b * Tt + t) * NWPAD + h * Cc;

  float sc[3];
  #pragma unroll
  for (int i = 0; i < 3; ++i) {
    const int c = ln + 32 * i;
    const int s = t + c - HALFc;
    const bool ok = (c < Cc) && (s >= 0) && (s < Tt);
    const int cc = (c < Cc) ? c : (Cc - 1);     // clamped, always-valid address
    const float w = wp[cc];
    sc[i] = ok ? w : NEGF;
  }
  float m = fmaxf(sc[0], fmaxf(sc[1], sc[2]));
  #pragma unroll
  for (int off = 1; off < 32; off <<= 1) m = fmaxf(m, __shfl_xor(m, off));

  float ps = 0.0f;
  #pragma unroll
  for (int i = 0; i < 3; ++i) {
    const int c = ln + 32 * i;
    const float p = (sc[i] > 0.5f * NEGF) ? __expf(sc[i] - m) : 0.0f;
    if (c < Cc) lp[wv][c] = p;
    ps += (c < Cc) ? p : 0.0f;
  }
  #pragma unroll
  for (int off = 1; off < 32; off <<= 1) ps += __shfl_xor(ps, off);
  asm volatile("s_wait_dscnt 0" ::: "memory");

  const float inv = 1.0f / ps;
  float a0 = 0.0f, a1 = 0.0f;
  const float* vb = V3 + (size_t)b * Tt * Ff + h * DKk;
  for (int c = 0; c < Cc; ++c) {
    const int s = t + c - HALFc;
    if ((unsigned)s < (unsigned)Tt) {
      const float a = lp[wv][c];
      const float* vr = vb + (size_t)s * Ff;
      a0 += a * vr[ln];
      a1 += a * vr[ln + 32];
    }
  }
  float* xo = X + ((size_t)b * Tt + t) * Ff + h * DKk;
  xo[ln]      = a0 * inv;
  xo[ln + 32] = a1 * inv;
}

// =====================================================================
// Flash-style fused MHA. One wave = 16 queries x all keys for one (b,h).
// 32-key chunks: 4 WMMA for S, online softmax, P via LDS transpose,
// V staged transposed in LDS, 4 WMMA for P@V.
// =====================================================================
__global__ __launch_bounds__(256)
void flash_mha(const float* __restrict__ Q, const float* __restrict__ Kin,
               const float* __restrict__ V, const int* __restrict__ mask,
               float* __restrict__ O)
{
  __shared__ __align__(16) __bf16 lpw[8][16 * 32];   // per-wave P tile (16 x 32)
  __shared__ __align__(16) __bf16 lvt[8][64 * 32];   // per-wave V^T tile (64 x 32)

  const int tid = threadIdx.x, wv = tid >> 5, ln = tid & 31;
  const int lmod = ln & 15, lhalf = ln >> 4;
  const int TQ = Tt / 16;                     // 128 query tiles
  const int wid = blockIdx.x * 8 + wv;
  const int qt = wid % TQ;
  const int h  = (wid / TQ) % Hh;
  const int b  = wid / (TQ * Hh);
  const int tq0 = qt * 16;
  const size_t bb = (size_t)b * Tt * Ff;
  const int hoff = h * DKk;
  const float scale = 0.125f;                 // 1/sqrt(64)

  __bf16* P  = lpw[wv];
  __bf16* VT = lvt[wv];

  // Q as two 16x32 A-fragments (dim chunks 0-31, 32-63)
  v16bf qa[2];
  {
    const int row = tq0 + lmod;
    const float* qp = Q + bb + (size_t)row * Ff + hoff + lhalf * 8;
    qa[0] = loadA16f_bf(qp);
    qa[1] = loadA16f_bf(qp + 32);
  }

  float rm[8], rs[8];
  #pragma unroll
  for (int r = 0; r < 8; ++r) { rm[r] = NEGF; rs[r] = 0.0f; }
  v8f o[4] = {};

  const int* mbase = mask + (size_t)b * Tt * Tt;

  for (int s0 = 0; s0 < Tt; s0 += 32) {
    // ---- S = Q @ K^T for two 16-key tiles ----
    v8f st[2];
    #pragma unroll
    for (int t2 = 0; t2 < 2; ++t2) {
      const int key = s0 + t2 * 16 + lmod;
      const float* kp = Kin + bb + (size_t)key * Ff + hoff + lhalf * 8;
      v16bf kb0 = loadA16f_bf(kp);        // dims 0-31 contraction
      v16bf kb1 = loadA16f_bf(kp + 32);   // dims 32-63 contraction
      v8f s = {};
      s = __builtin_amdgcn_wmma_f32_16x16x32_bf16(false, qa[0], false, kb0,
                                                  (short)0, s, false, false);
      s = __builtin_amdgcn_wmma_f32_16x16x32_bf16(false, qa[1], false, kb1,
                                                  (short)0, s, false, false);
      st[t2] = s;
    }

    // ---- mask + online softmax (row stats shared within lane-half) ----
    const int k0 = s0 + lmod, k1 = s0 + 16 + lmod;
    #pragma unroll
    for (int r = 0; r < 8; ++r) {
      const int row = tq0 + r + 8 * lhalf;
      float s0v = st[0][r] * scale;
      float s1v = st[1][r] * scale;
      if (mbase[(size_t)row * Tt + k0] == 0) s0v = NEGF;
      if (mbase[(size_t)row * Tt + k1] == 0) s1v = NEGF;
      float v = fmaxf(s0v, s1v);
      #pragma unroll
      for (int off = 1; off < 16; off <<= 1) v = fmaxf(v, __shfl_xor(v, off));
      const float mnew = fmaxf(rm[r], v);
      const float corr = __expf(rm[r] - mnew);
      const float p0 = (s0v > 0.5f * NEGF) ? __expf(s0v - mnew) : 0.0f;
      const float p1 = (s1v > 0.5f * NEGF) ? __expf(s1v - mnew) : 0.0f;
      float psum = p0 + p1;
      #pragma unroll
      for (int off = 1; off < 16; off <<= 1) psum += __shfl_xor(psum, off);
      rs[r] = rs[r] * corr + psum;
      rm[r] = mnew;
      #pragma unroll
      for (int dc = 0; dc < 4; ++dc) o[dc][r] *= corr;
      // scatter P (C layout -> LDS row-major 16x32)
      P[(r + 8 * lhalf) * 32 + lmod]      = (__bf16)p0;
      P[(r + 8 * lhalf) * 32 + 16 + lmod] = (__bf16)p1;
    }

    // ---- stage V chunk transposed: VT[dim][key], lane owns 2 dims ----
    {
      const int d0 = ln * 2;
      const float* vp = V + bb + hoff + d0;
      __align__(16) __bf16 row0[32];
      __align__(16) __bf16 row1[32];
      #pragma unroll
      for (int j = 0; j < 32; ++j) {
        float2 vv = *(const float2*)(vp + (size_t)(s0 + j) * Ff);
        row0[j] = (__bf16)vv.x;
        row1[j] = (__bf16)vv.y;
      }
      #pragma unroll
      for (int q = 0; q < 4; ++q) {
        *(v8bf*)&VT[d0 * 32 + q * 8]       = *(const v8bf*)&row0[q * 8];
        *(v8bf*)&VT[(d0 + 1) * 32 + q * 8] = *(const v8bf*)&row1[q * 8];
      }
    }
    // cross-lane LDS transpose hazard: make stores visible before frag loads
    asm volatile("s_wait_dscnt 0" ::: "memory");

    // ---- O += P @ V  (A-frag from P, B-frags from VT) ----
    v16bf pa;
    {
      const int kb = lhalf * 8;
      v8bf plo = *(const v8bf*)&P[lmod * 32 + kb];
      v8bf phi = *(const v8bf*)&P[lmod * 32 + kb + 16];
      pa = cat8(plo, phi);
    }
    #pragma unroll
    for (int dc = 0; dc < 4; ++dc) {
      const int dim = dc * 16 + lmod;
      v8bf blo = *(const v8bf*)&VT[dim * 32 + lhalf * 8];
      v8bf bhi = *(const v8bf*)&VT[dim * 32 + lhalf * 8 + 16];
      v16bf vb = cat8(blo, bhi);
      o[dc] = __builtin_amdgcn_wmma_f32_16x16x32_bf16(false, pa, false, vb,
                                                      (short)0, o[dc], false, false);
    }
  }

  // ---- normalize and write (B,T,F) ----
  #pragma unroll
  for (int dc = 0; dc < 4; ++dc) {
    const int col = hoff + dc * 16 + lmod;
    #pragma unroll
    for (int r = 0; r < 8; ++r) {
      const int row = tq0 + r + 8 * lhalf;
      const float d = rs[r];
      const float val = (d > 0.0f) ? (o[dc][r] / d) : 0.0f;
      O[bb + (size_t)row * Ff + col] = val;
    }
  }
}

// =====================================================================
// Host-side orchestration
// =====================================================================
extern "C" void kernel_launch(void* const* d_in, const int* in_sizes, int n_in,
                              void* d_out, int out_size, void* d_ws, size_t ws_size,
                              hipStream_t stream) {
  (void)in_sizes; (void)n_in; (void)out_size; (void)ws_size;
  const float* query  = (const float*)d_in[0];
  /* d_in[1] = key: unused by the reference computation */
  const float* value  = (const float*)d_in[2];
  const int*   mask   = (const int*)  d_in[3];
  const float* w1     = (const float*)d_in[4];
  const float* w2     = (const float*)d_in[5];
  const float* w3     = (const float*)d_in[6];
  const float* w_ldsa = (const float*)d_in[7];
  const float* wq     = (const float*)d_in[8];
  const float* bq     = (const float*)d_in[9];
  const float* wk     = (const float*)d_in[10];
  const float* bk     = (const float*)d_in[11];
  const float* wvv    = (const float*)d_in[12];
  const float* bv     = (const float*)d_in[13];
  const float* wo     = (const float*)d_in[14];
  const float* bo     = (const float*)d_in[15];
  float* out = (float*)d_out;

  const int M = Bb * Tt;                    // 8192
  const size_t NF = (size_t)M * Ff;         // 2,097,152 floats
  const size_t NWP = (size_t)M * NWPAD;     // 3,145,728 floats (padded)

  float* bufA = (float*)d_ws;               // NF   (q_relu -> v3 -> qh)
  float* bufB = bufA + NF;                  // NWP  (band weights -> kh)
  float* bufC = bufB + NWP;                 // NF   (ldsa x -> vh)
  float* bufD = bufC + NF;                  // NF   (x2 -> attn_out)
  float* w2p  = bufD + NF;                  // 256 x 384 padded w2

  const dim3 blk(256);
  const int MT = M / 64;                    // 128

  // 0) pad w2 into 256x384
  pad_w2<<<dim3((Ff * NWPAD) / 256), blk, 0, stream>>>(w2, w2p);
  // 1) q_relu = relu(query @ w1)
  gemm_bf16_wmma<<<dim3(MT, Ff / 128), blk, 0, stream>>>(query, w1, nullptr, bufA, M, Ff, Ff, 1);
  // 2) band weights = q_relu @ w2p   (N = 384 padded)
  gemm_bf16_wmma<<<dim3(MT, NWPAD / 128), blk, 0, stream>>>(bufA, w2p, nullptr, bufB, M, NWPAD, Ff, 0);
  // 3) v3 = value @ w3
  gemm_bf16_wmma<<<dim3(MT, Ff / 128), blk, 0, stream>>>(value, w3, nullptr, bufA, M, Ff, Ff, 0);
  // 4) banded local attention -> bufC
  ldsa_band<<<dim3((Bb * Hh * Tt) / 8), blk, 0, stream>>>(bufB, bufA, bufC);
  // 5) x2 = x @ w_ldsa
  gemm_bf16_wmma<<<dim3(MT, Ff / 128), blk, 0, stream>>>(bufC, w_ldsa, nullptr, bufD, M, Ff, Ff, 0);
  // 6) MHA projections
  gemm_bf16_wmma<<<dim3(MT, Ff / 128), blk, 0, stream>>>(bufD, wq, bq, bufA, M, Ff, Ff, 0);
  gemm_bf16_wmma<<<dim3(MT, Ff / 128), blk, 0, stream>>>(bufD, wk, bk, bufB, M, Ff, Ff, 0);
  gemm_bf16_wmma<<<dim3(MT, Ff / 128), blk, 0, stream>>>(bufD, wvv, bv, bufC, M, Ff, Ff, 0);
  // 7) fused flash attention -> bufD
  flash_mha<<<dim3((Bb * Hh * (Tt / 16)) / 8), blk, 0, stream>>>(bufA, bufB, bufC, mask, bufD);
  // 8) out = attn_out @ wo + bo
  gemm_bf16_wmma<<<dim3(MT, Ff / 128), blk, 0, stream>>>(bufD, wo, bo, out, M, Ff, Ff, 0);
}
